// GNNSegmentClassifier_26182120636657
// MI455X (gfx1250) — matrix operand order repaired and checked by reading, not compiled, not run on hardware
//
#include <hip/hip_runtime.h>
#include <hip/hip_bf16.h>
#include <math.h>

// ---- CDNA5 vector types -----------------------------------------------------
typedef __attribute__((ext_vector_type(16))) _Float16 v16h;
typedef __attribute__((ext_vector_type(8)))  _Float16 v8h;
typedef __attribute__((ext_vector_type(8)))  float    v8f;

#define DPAD 12   // latent dim 11 padded to 12 floats (48B rows, float4-aligned)

// Hardware transcendentals (gfx1250 has V_TANH_F32; V_EXP_F32 is base-2).
__device__ __forceinline__ float fast_tanh(float x) {
#if __has_builtin(__builtin_amdgcn_tanhf)
    return __builtin_amdgcn_tanhf(x);
#else
    return tanhf(x);
#endif
}

__device__ __forceinline__ float fast_sigmoid(float x) {
#if __has_builtin(__builtin_amdgcn_exp2f) && __has_builtin(__builtin_amdgcn_rcpf)
    // sigmoid(x) = 1 / (1 + 2^(-x*log2(e)))
    return __builtin_amdgcn_rcpf(1.0f + __builtin_amdgcn_exp2f(-1.442695041f * x));
#else
    return 1.0f / (1.0f + expf(-x));
#endif
}

// ---- node init: H = tanh(x @ W_in + b_in); h = [H, x, 0] --------------------
__global__ __launch_bounds__(256) void gnn_init_nodes(
    const float* __restrict__ x, const float* __restrict__ W_in,
    const float* __restrict__ b_in, float* __restrict__ h, int N)
{
    int nid = blockIdx.x * 256 + threadIdx.x;
    if (nid >= N) return;
    float x0 = x[(size_t)nid * 3 + 0];
    float x1 = x[(size_t)nid * 3 + 1];
    float x2 = x[(size_t)nid * 3 + 2];
    float Hj[8];
#pragma unroll
    for (int j = 0; j < 8; ++j)
        Hj[j] = fast_tanh(b_in[j] + x0 * W_in[j] + x1 * W_in[8 + j] + x2 * W_in[16 + j]);
    float4* hp = (float4*)(h + (size_t)nid * DPAD);
    hp[0] = make_float4(Hj[0], Hj[1], Hj[2], Hj[3]);
    hp[1] = make_float4(Hj[4], Hj[5], Hj[6], Hj[7]);
    hp[2] = make_float4(x0, x1, x2, 0.0f);
}

// ---- zero the message accumulators (only needed once up front) --------------
__global__ __launch_bounds__(256) void gnn_zero_f32(float* __restrict__ p, int n)
{
    int i = blockIdx.x * 256 + threadIdx.x;
    if (i < n) p[i] = 0.0f;
}

// ---- edge pass: e = sigmoid(tanh([h_c,h_r] @ We1 + be1) @ We2 + be2) --------
// layer-1 GEMM done with v_wmma_f32_16x16x32_f16 on 16-edge tiles; then
// scatter-add e*h into mi[col]/mo[row], or store e on the final pass.
__global__ __launch_bounds__(256) void gnn_edge_pass(
    const float* __restrict__ h, float* __restrict__ mi, float* __restrict__ mo,
    const int* __restrict__ rowp, const int* __restrict__ colp,
    const float* __restrict__ We1, const float* __restrict__ be1,
    const float* __restrict__ We2p, const float* __restrict__ be2p,
    int E, float* __restrict__ out)
{
    // per-wave staging: 32 edges x 32 K-slots (f16) in the 16-bit A layout
    __shared__ __attribute__((aligned(16))) _Float16 sA[8][32 * 32];
    __shared__ float sE[8][32];

    const int tid  = threadIdx.x;
    const int w    = tid >> 5;
    const int lane = tid & 31;
    const int ncol = lane & 15;            // B/D column this lane owns
    const int kbB  = (lane < 16) ? 0 : 16; // B fragment K-base

    // B fragment = We1 (22x8), columns >=8 and K>=22 zero-padded
    v16h bfrag;
#pragma unroll
    for (int j = 0; j < 16; ++j) {
        int k = kbB + j;
        float wv = (ncol < 8 && k < 22) ? We1[k * 8 + ncol] : 0.0f;
        bfrag[j] = (_Float16)wv;
    }
    const float be1n = (ncol < 8) ? be1[ncol]  : 0.0f;
    const float we2n = (ncol < 8) ? We2p[ncol] : 0.0f;
    const float be2  = be2p[0];

    const int  e_id  = blockIdx.x * 256 + tid;
    const bool valid = (e_id < E);
    const int  ec    = valid ? e_id : (E - 1);

    // stream the index arrays ahead (global_prefetch_b8)
    __builtin_prefetch(rowp + e_id + 4096, 0, 0);
    __builtin_prefetch(colp + e_id + 4096, 0, 0);

    const int r = rowp[ec];
    const int c = colp[ec];

    union F12 { float4 v[3]; float f[12]; };
    F12 hc, hr;
    {
        const float4* pc = (const float4*)(h + (size_t)c * DPAD);
        const float4* pr = (const float4*)(h + (size_t)r * DPAD);
        hc.v[0] = pc[0]; hc.v[1] = pc[1]; hc.v[2] = pc[2];
        hr.v[0] = pr[0]; hr.v[1] = pr[1]; hr.v[2] = pr[2];
    }

    // stage this edge's feature row [h[col](11), h[row](11), pad] as f16
    union ST { v8h q[4]; _Float16 f[32]; } st;
#pragma unroll
    for (int k = 0; k < 11; ++k) st.f[k]      = (_Float16)hc.f[k];
#pragma unroll
    for (int k = 0; k < 11; ++k) st.f[11 + k] = (_Float16)hr.f[k];
#pragma unroll
    for (int k = 22; k < 32; ++k) st.f[k]     = (_Float16)0.0f;
    {
        v8h* dst = (v8h*)&sA[w][lane * 32];
        dst[0] = st.q[0]; dst[1] = st.q[1]; dst[2] = st.q[2]; dst[3] = st.q[3];
    }
    asm volatile("s_wait_dscnt 0" ::: "memory");

    // A fragments in contiguous K order, matching bfrag:
    // lanes 0-15 hold K {0..15}, lanes 16-31 hold K {16..31}
    const int klo  = (lane < 16) ? 0 : 16;
    const int arow = lane & 15;
    union AV { v8h q[2]; v16h v; };
    AV a0, a1;
    a0.q[0] = *(const v8h*)&sA[w][arow * 32 + klo];
    a0.q[1] = *(const v8h*)&sA[w][arow * 32 + klo + 8];
    a1.q[0] = *(const v8h*)&sA[w][(16 + arow) * 32 + klo];
    a1.q[1] = *(const v8h*)&sA[w][(16 + arow) * 32 + klo + 8];

    v8f acc0 = {}, acc1 = {};
    acc0 = __builtin_amdgcn_wmma_f32_16x16x32_f16(false, a0.v, false, bfrag,
                                                  (short)0, acc0, false, false);
    acc1 = __builtin_amdgcn_wmma_f32_16x16x32_f16(false, a1.v, false, bfrag,
                                                  (short)0, acc1, false, false);

    // per-column tanh + We2 weighting, then sum across hidden units.
    // Columns 8..15 are exact zeros (zero-padded B / be1 / We2), so reducing
    // over lanes n=0..7 of each 16-lane group suffices: 3 butterfly stages,
    // and the writer lanes (0 and 16) are inside that closure.
    float g0[8], g1[8];
#pragma unroll
    for (int rr = 0; rr < 8; ++rr) {
        float s0 = fast_tanh(acc0[rr] + be1n) * we2n;
        float s1 = fast_tanh(acc1[rr] + be1n) * we2n;
        s0 += __shfl_xor(s0, 1, 16);  s1 += __shfl_xor(s1, 1, 16);
        s0 += __shfl_xor(s0, 2, 16);  s1 += __shfl_xor(s1, 2, 16);
        s0 += __shfl_xor(s0, 4, 16);  s1 += __shfl_xor(s1, 4, 16);
        g0[rr] = s0 + be2;
        g1[rr] = s1 + be2;
    }

    // lane 0 holds rows 0..7, lane 16 holds rows 8..15 of each tile (logits)
    if (lane == 0) {
#pragma unroll
        for (int rr = 0; rr < 8; ++rr) {
            sE[w][rr]      = g0[rr];
            sE[w][16 + rr] = g1[rr];
        }
    }
    if (lane == 16) {
#pragma unroll
        for (int rr = 0; rr < 8; ++rr) {
            sE[w][8 + rr]  = g0[rr];
            sE[w][24 + rr] = g1[rr];
        }
    }
    asm volatile("s_wait_dscnt 0" ::: "memory");
    const float ev = fast_sigmoid(sE[w][lane]);   // each lane gates its own edge

    if (out) {                       // final pass: emit gate per edge
        if (valid) out[e_id] = ev;
    } else if (valid) {              // message pass: scatter-add
        float* mib = mi + (size_t)c * DPAD;   // mi[col] += e * h[row]
        float* mob = mo + (size_t)r * DPAD;   // mo[row] += e * h[col]
#pragma unroll
        for (int k = 0; k < 11; ++k) {
            atomicAdd(mib + k, ev * hr.f[k]);
            atomicAdd(mob + k, ev * hc.f[k]);
        }
    }
}

// ---- node pass: H = tanh(tanh([mi,mo,h] @ Wn1 + bn1) @ Wn2 + bn2) ----------
// Also re-zeroes this node's mi/mo rows for the next iteration's edge pass.
__global__ __launch_bounds__(256) void gnn_node_pass(
    float* __restrict__ h, float* __restrict__ mi, float* __restrict__ mo,
    const float* __restrict__ Wn1, const float* __restrict__ bn1,
    const float* __restrict__ Wn2, const float* __restrict__ bn2, int N)
{
    __shared__ __attribute__((aligned(16))) float sW1[33 * 8];
    __shared__ __attribute__((aligned(16))) float sW2[64];
    __shared__ float sB1[8];
    __shared__ float sB2[8];

    const int tid = threadIdx.x;
    if (tid < 264) sW1[tid] = Wn1[tid];
    if (tid >= 256 - 64) sW2[tid - (256 - 64)] = Wn2[tid - (256 - 64)];
    if (tid < 8) { sB1[tid] = bn1[tid]; sB2[tid] = bn2[tid]; }
    __syncthreads();

    const int nid = blockIdx.x * 256 + tid;
    if (nid >= N) return;

    union F12 { float4 v[3]; float f[12]; };
    F12 vmi, vmo, vh;
    {
        const float4* p;
        p = (const float4*)(mi + (size_t)nid * DPAD); vmi.v[0] = p[0]; vmi.v[1] = p[1]; vmi.v[2] = p[2];
        p = (const float4*)(mo + (size_t)nid * DPAD); vmo.v[0] = p[0]; vmo.v[1] = p[1]; vmo.v[2] = p[2];
        p = (const float4*)(h  + (size_t)nid * DPAD); vh.v[0]  = p[0]; vh.v[1]  = p[1]; vh.v[2]  = p[2];
    }
    float feat[33];
#pragma unroll
    for (int k = 0; k < 11; ++k) {
        feat[k]      = vmi.f[k];
        feat[11 + k] = vmo.f[k];
        feat[22 + k] = vh.f[k];
    }

    // re-zero accumulators for the next iteration (fused, saves a full pass)
    {
        const float4 z = make_float4(0.f, 0.f, 0.f, 0.f);
        float4* pz;
        pz = (float4*)(mi + (size_t)nid * DPAD); pz[0] = z; pz[1] = z; pz[2] = z;
        pz = (float4*)(mo + (size_t)nid * DPAD); pz[0] = z; pz[1] = z; pz[2] = z;
    }

    float t1[8];
#pragma unroll
    for (int j = 0; j < 8; ++j) t1[j] = sB1[j];
#pragma unroll
    for (int k = 0; k < 33; ++k) {
        const float f = feat[k];
        const float4* wr = (const float4*)&sW1[k * 8];
        const float4 wa = wr[0], wb = wr[1];
        t1[0] += f * wa.x; t1[1] += f * wa.y; t1[2] += f * wa.z; t1[3] += f * wa.w;
        t1[4] += f * wb.x; t1[5] += f * wb.y; t1[6] += f * wb.z; t1[7] += f * wb.w;
    }
    float a1[8];
#pragma unroll
    for (int j = 0; j < 8; ++j) a1[j] = fast_tanh(t1[j]);

    float t2[8];
#pragma unroll
    for (int j = 0; j < 8; ++j) t2[j] = sB2[j];
#pragma unroll
    for (int i = 0; i < 8; ++i) {
        const float f = a1[i];
        const float4* wr = (const float4*)&sW2[i * 8];
        const float4 wa = wr[0], wb = wr[1];
        t2[0] += f * wa.x; t2[1] += f * wa.y; t2[2] += f * wa.z; t2[3] += f * wa.w;
        t2[4] += f * wb.x; t2[5] += f * wb.y; t2[6] += f * wb.z; t2[7] += f * wb.w;
    }
    float4* hp = (float4*)(h + (size_t)nid * DPAD);
    hp[0] = make_float4(fast_tanh(t2[0]), fast_tanh(t2[1]), fast_tanh(t2[2]), fast_tanh(t2[3]));
    hp[1] = make_float4(fast_tanh(t2[4]), fast_tanh(t2[5]), fast_tanh(t2[6]), fast_tanh(t2[7]));
    // hp[2] (the x part + pad) stays untouched
}

// ---- launch -----------------------------------------------------------------
extern "C" void kernel_launch(void* const* d_in, const int* in_sizes, int n_in,
                              void* d_out, int out_size, void* d_ws, size_t ws_size,
                              hipStream_t stream)
{
    const float* x    = (const float*)d_in[0];
    const int*   eidx = (const int*)  d_in[1];
    const float* W_in = (const float*)d_in[2];
    const float* b_in = (const float*)d_in[3];
    const float* We1  = (const float*)d_in[4];
    const float* be1  = (const float*)d_in[5];
    const float* We2  = (const float*)d_in[6];
    const float* be2  = (const float*)d_in[7];
    const float* Wn1  = (const float*)d_in[8];
    const float* bn1  = (const float*)d_in[9];
    const float* Wn2  = (const float*)d_in[10];
    const float* bn2  = (const float*)d_in[11];

    const int N = in_sizes[0] / 3;
    const int E = in_sizes[1] / 2;
    const int* rowp = eidx;        // edge_index[0]
    const int* colp = eidx + E;    // edge_index[1]

    // workspace: h | mi | mo, each N x 12 f32  (14.4 MB total)
    float* h  = (float*)d_ws;
    float* mi = h  + (size_t)N * DPAD;
    float* mo = mi + (size_t)N * DPAD;

    const int nb = (N + 255) / 256;
    const int eb = (E + 255) / 256;

    gnn_init_nodes<<<nb, 256, 0, stream>>>(x, W_in, b_in, h, N);

    // zero mi|mo once; node_pass re-zeroes them each iteration after consuming
    const int zn = N * DPAD * 2;
    gnn_zero_f32<<<(zn + 255) / 256, 256, 0, stream>>>(mi, zn);

    for (int it = 0; it < 3; ++it) {
        gnn_edge_pass<<<eb, 256, 0, stream>>>(h, mi, mo, rowp, colp,
                                              We1, be1, We2, be2, E, nullptr);
        gnn_node_pass<<<nb, 256, 0, stream>>>(h, mi, mo, Wn1, bn1, Wn2, bn2, N);
    }
    gnn_edge_pass<<<eb, 256, 0, stream>>>(h, mi, mo, rowp, colp,
                                          We1, be1, We2, be2, E, (float*)d_out);
}
